// MSVA_Gen_auto_17377437680003
// MI455X (gfx1250) — matrix operand order repaired and checked by reading, not compiled, not run on hardware
//
#include <hip/hip_runtime.h>
#include <math.h>

// ---------------------------------------------------------------------------
// MSVA (multi-stream banded self-attention) for MI455X / gfx1250.
// All GEMMs use v_wmma_f32_16x16x32_bf16 (fp32 accumulate). Activations and
// weights converted to bf16 ONCE (weights transposed to [N][K]) so fragments
// are contiguous b128 loads. GEMM: 4-wave blocks, B tile staged through a
// double-buffered LDS ring via global_load_async_to_lds_b128 (ASYNCcnt),
// A fragments double-buffered in *named* registers (no dynamic indexing so
// they stay in VGPRs), k-loop unrolled x2 so buffer indices are literals.
// Band structure (|i-j| <= 249) cuts attention work ~8x vs dense.
// ---------------------------------------------------------------------------

#define N_FRAMES 4096
#define D_FEAT   1024
#define KEEP     249          // |i-j| <= KEEP kept (APPERTURE-1)
#define BSPAN    528          // columns covered per 16-row block: [j0-256, j0+272)
#define NTILES   33           // BSPAN/16
#define LN_EPS   1e-6f

#if __has_builtin(__builtin_amdgcn_global_load_async_to_lds_b128) && \
    __has_builtin(__builtin_amdgcn_s_wait_asynccnt)
#define ASYNC_LDS 1
#endif

typedef __attribute__((ext_vector_type(16))) __bf16 v16bf;
typedef __attribute__((ext_vector_type(8)))  float  v8f;
typedef __attribute__((ext_vector_type(4)))  int    int4v;
typedef unsigned short u16;

#ifdef ASYNC_LDS
typedef __attribute__((address_space(1))) int4v gas_int4;
typedef __attribute__((address_space(3))) int4v las_int4;
#endif

union FragU {
    v16bf v;
    uint4 q[2];
    u16   s[16];
};

__device__ __forceinline__ u16 f2bf_bits(float f) {
    unsigned u = __float_as_uint(f);
    return (u16)((u + 0x7FFFu + ((u >> 16) & 1u)) >> 16);   // RNE
}

__device__ __forceinline__ v8f wmma_bf16(const v16bf& a, const v16bf& b, const v8f& c) {
    return __builtin_amdgcn_wmma_f32_16x16x32_bf16(false, a, false, b, (short)0, c, false, false);
}

// Copy 16 bf16 (32B) global -> LDS; ASYNCcnt-tracked when available.
__device__ __forceinline__ void stage16(const u16* g, u16* l) {
#ifdef ASYNC_LDS
    __builtin_amdgcn_global_load_async_to_lds_b128(
        (gas_int4*)g, (las_int4*)l, 0, 0);
    __builtin_amdgcn_global_load_async_to_lds_b128(
        (gas_int4*)(g + 8), (las_int4*)(l + 8), 0, 0);
#else
    *(uint4*)l       = *(const uint4*)g;
    *(uint4*)(l + 8) = *(const uint4*)(g + 8);
#endif
}

__device__ __forceinline__ void wait_async_zero() {
#ifdef ASYNC_LDS
    __builtin_amdgcn_s_wait_asynccnt(0);
#endif
}

// A fragment (16x32, MxK) from bf16 row-major:
// lane L: M=L&15; chunks at k0+kb and k0+kb+16, kb=(L>=16)?8:0.
__device__ __forceinline__ void load_a16(const u16* __restrict__ A, int lda,
                                         int m0, int k0, int lane, FragU& a) {
    const u16* p = A + (size_t)(m0 + (lane & 15)) * lda + k0 + ((lane >= 16) ? 8 : 0);
    a.q[0] = *(const uint4*)p;
    a.q[1] = *(const uint4*)(p + 16);
}

// ---------------------------------------------------------------------------
__global__ __launch_bounds__(256)
void cvt_bf16_kernel(const float* __restrict__ src, u16* __restrict__ dst, size_t count) {
    size_t i = (size_t)blockIdx.x * blockDim.x + threadIdx.x;
    size_t stride = (size_t)gridDim.x * blockDim.x;
    for (; i < count; i += stride) dst[i] = f2bf_bits(src[i]);
}

// fp32 [R][C] -> bf16 transposed [C][R]
__global__ __launch_bounds__(256)
void cvt_bf16_t_kernel(const float* __restrict__ src, u16* __restrict__ dst, int R, int C) {
    size_t i = (size_t)blockIdx.x * blockDim.x + threadIdx.x;
    size_t total = (size_t)R * C;
    size_t stride = (size_t)gridDim.x * blockDim.x;
    for (; i < total; i += stride) {
        int r = (int)(i / C), c = (int)(i - (size_t)r * C);
        dst[(size_t)c * R + r] = f2bf_bits(src[i]);
    }
}

// ---------------------------------------------------------------------------
// GEMM: C[M,N] = act(alpha * A @ B + bias). A bf16 row-major, B as bf16 B^T.
// Block = 4 waves = 64x64 tile (each wave a 16-row strip over 4 N-subtiles).
// Requires K % 64 == 0 (true for all uses: 1024, 512).
// ---------------------------------------------------------------------------
#define BSTRIDE 40   // LDS row pitch in halfs (80B) -> conflict-spread, 16B aligned

// Load all 4 B fragments (8 ds_load_b128 in a batch), then 4 WMMAs.
__device__ __forceinline__ void mma4(const u16* __restrict__ bbase, int lane, int kbB,
                                     const FragU& a, v8f acc[4]) {
    const u16* r0 = bbase + (size_t)((lane & 15) +  0) * BSTRIDE + kbB;
    const u16* r1 = bbase + (size_t)((lane & 15) + 16) * BSTRIDE + kbB;
    const u16* r2 = bbase + (size_t)((lane & 15) + 32) * BSTRIDE + kbB;
    const u16* r3 = bbase + (size_t)((lane & 15) + 48) * BSTRIDE + kbB;
    FragU b0, b1, b2, b3;
    b0.q[0] = *(const uint4*)r0;  b0.q[1] = *(const uint4*)(r0 + 8);
    b1.q[0] = *(const uint4*)r1;  b1.q[1] = *(const uint4*)(r1 + 8);
    b2.q[0] = *(const uint4*)r2;  b2.q[1] = *(const uint4*)(r2 + 8);
    b3.q[0] = *(const uint4*)r3;  b3.q[1] = *(const uint4*)(r3 + 8);
    acc[0] = wmma_bf16(a.v, b0.v, acc[0]);
    acc[1] = wmma_bf16(a.v, b1.v, acc[1]);
    acc[2] = wmma_bf16(a.v, b2.v, acc[2]);
    acc[3] = wmma_bf16(a.v, b3.v, acc[3]);
}

__global__ __launch_bounds__(128)
void gemm_bf16_kernel(const u16* __restrict__ A, const u16* __restrict__ Bt,
                      const float* __restrict__ bias,
                      float* __restrict__ Cf, u16* __restrict__ Cb, u16* __restrict__ Cbt,
                      int M, int N, int K, float alpha, int relu) {
    __shared__ u16 Bs0[64 * BSTRIDE];
    __shared__ u16 Bs1[64 * BSTRIDE];
    int n0 = blockIdx.x * 64;
    int m0 = blockIdx.y * 64;
    int tid = threadIdx.x;
    int wave = tid >> 5;
    int lane = tid & 31;
    int m0w = m0 + wave * 16;

    // staging: thread t copies 16 halfs of B^T row (n0 + t/2), half (t&1)
    int snl = tid >> 1;
    int sh = (tid & 1) * 16;
    const u16* sgl = Bt + (size_t)(n0 + snl) * K + sh;
    u16* sl0 = &Bs0[snl * BSTRIDE + sh];
    u16* sl1 = &Bs1[snl * BSTRIDE + sh];

    const u16* arow = A + (size_t)(m0w + (lane & 15)) * K + ((lane >= 16) ? 16 : 0);
    int kbB = (lane >= 16) ? 16 : 0;

    v8f acc[4] = {{}, {}, {}, {}};
    FragU a0, a1;

    stage16(sgl, sl0);                          // prologue: slice k=0 -> buf 0
    a0.q[0] = *(const uint4*)(arow + 0);
    a0.q[1] = *(const uint4*)(arow + 8);

    for (int k0 = 0; k0 < K; k0 += 64) {
        // ---- phase 0: consume buf0 / a0, stage buf1 / a1 (k0+32 always < K)
        wait_async_zero();
        __syncthreads();
        stage16(sgl + k0 + 32, sl1);
        a1.q[0] = *(const uint4*)(arow + k0 + 32);
        a1.q[1] = *(const uint4*)(arow + k0 + 40);
        mma4(Bs0, lane, kbB, a0, acc);

        // ---- phase 1: consume buf1 / a1, stage buf0 / a0 for k0+64
        wait_async_zero();
        __syncthreads();
        if (k0 + 64 < K) {
            stage16(sgl + k0 + 64, sl0);
            a0.q[0] = *(const uint4*)(arow + k0 + 64);
            a0.q[1] = *(const uint4*)(arow + k0 + 72);
        }
        mma4(Bs1, lane, kbB, a1, acc);
    }

    int hb = (lane >= 16) ? 8 : 0;
#pragma unroll
    for (int i = 0; i < 4; ++i) {
        int n = n0 + 16 * i + (lane & 15);
        float bv = bias ? bias[n] : 0.0f;
#pragma unroll
        for (int r = 0; r < 8; ++r) {
            float v = acc[i][r] * alpha + bv;
            if (relu) v = fmaxf(v, 0.0f);
            int m = m0w + r + hb;
            if (Cf)  Cf[(size_t)m * N + n]  = v;
            if (Cb)  Cb[(size_t)m * N + n]  = f2bf_bits(v);
            if (Cbt) Cbt[(size_t)n * M + m] = f2bf_bits(v);
        }
    }
}

// ---------------------------------------------------------------------------
// Banded logits + masked softmax. One block (4 waves) per 16 rows.
// Writes w^T as bf16 (WtB[i][j] = w[j][i], row-major -> contiguous for attn_y)
// and optionally the fp32 full w (last stream -> d_out att region).
// ---------------------------------------------------------------------------
__global__ __launch_bounds__(128)
void attn_band_kernel(const u16* __restrict__ Qb, const u16* __restrict__ Kb,
                      float* __restrict__ Wf, u16* __restrict__ WtB, int n, int d) {
    __shared__ float lbuf[16][BSPAN];
    __shared__ float red[16][8];
    __shared__ float rowm[16];
    __shared__ float rowd[16];

    int j0 = blockIdx.x * 16;
    int tid = threadIdx.x;
    int wave = tid >> 5;
    int lane = tid & 31;

    for (int t = wave; t < NTILES; t += 4) {
        int c0 = j0 - 256 + t * 16;
        v8f acc = {};
        bool any = (c0 + 15 >= 0) && (c0 < n);          // uniform per wave
        if (any) {
            int c = c0 + (lane & 15);
            int cc = c < 0 ? 0 : (c >= n ? n - 1 : c);
            const u16* brow = Kb + (size_t)cc * d + ((lane >= 16) ? 8 : 0);
#pragma unroll 2
            for (int k0 = 0; k0 < d; k0 += 32) {
                if (k0 + 32 < d) __builtin_prefetch(brow + k0 + 32, 0, 1);
                FragU a, b;
                load_a16(Qb, d, j0, k0, lane, a);
                b.q[0] = *(const uint4*)(brow + k0);
                b.q[1] = *(const uint4*)(brow + k0 + 16);
                acc = wmma_bf16(a.v, b.v, acc);
            }
        }
        int hb = (lane >= 16) ? 8 : 0;
        int c = c0 + (lane & 15);
#pragma unroll
        for (int r = 0; r < 8; ++r) {
            int row = r + hb;
            int j = j0 + row;
            int dd = j - c; if (dd < 0) dd = -dd;
            bool ok = any && (c >= 0) && (c < n) && (dd <= KEEP);
            lbuf[row][t * 16 + (lane & 15)] = ok ? acc[r] : -3.0e38f;
        }
    }
    __syncthreads();

    int r = tid & 15;
    int ch = tid >> 4;                                  // 8 chunks of 66
    float m = -3.4e38f;
    for (int q = ch * 66; q < (ch + 1) * 66; ++q) m = fmaxf(m, lbuf[r][q]);
    red[r][ch] = m;
    __syncthreads();
    if (tid < 16) {
        float mm = red[tid][0];
#pragma unroll
        for (int c2 = 1; c2 < 8; ++c2) mm = fmaxf(mm, red[tid][c2]);
        rowm[tid] = mm;
    }
    __syncthreads();
    float mm = rowm[r];
    float s = 0.0f;
    for (int q = ch * 66; q < (ch + 1) * 66; ++q) s += expf(lbuf[r][q] - mm);
    red[r][ch] = s;
    __syncthreads();
    if (tid < 16) {
        float ss = 0.0f;
#pragma unroll
        for (int c2 = 0; c2 < 8; ++c2) ss += red[tid][c2];
        rowd[tid] = ss;
    }
    __syncthreads();

    for (int idx = tid; idx < 16 * BSPAN; idx += 128) {
        int rr = idx / BSPAN;
        int q = idx - rr * BSPAN;
        int c = j0 - 256 + q;
        if (c >= 0 && c < n) {
            float wv = expf(lbuf[rr][q] - rowm[rr]) / rowd[rr];
            if (Wf) Wf[(size_t)(j0 + rr) * n + c] = wv;
            WtB[(size_t)c * n + (j0 + rr)] = f2bf_bits(wv);
        }
    }
}

// ---------------------------------------------------------------------------
// y = w^T @ V (banded). A = WtB (bf16 row-major, contiguous); B = V^T bf16.
// One wave per 16x16 tile, 17 K-tiles of 32. Edge tiles take guarded path.
// ---------------------------------------------------------------------------
__global__ __launch_bounds__(32)
void attn_y_kernel(const u16* __restrict__ WtB, const u16* __restrict__ Vt,
                   u16* __restrict__ Yb, int n, int d) {
    int n0 = blockIdx.x * 16;
    int i0 = blockIdx.y * 16;
    int lane = threadIdx.x;
    v8f acc = {};

    const u16* ap = WtB + (size_t)(i0 + (lane & 15)) * n;   // A[m][k] = WtB[i0+m][j0+k]
    const u16* bp = Vt + (size_t)(n0 + (lane & 15)) * n;    // B[k][nn] = Vt[n0+nn][j0+k]
    int kbA = (lane >= 16) ? 8 : 0;
    int kbB = (lane >= 16) ? 8 : 0;

    for (int j0 = i0 - 256; j0 <= i0 + 256; j0 += 32) {
        if (j0 + 31 < 0 || j0 >= n) continue;               // uniform
        FragU a, b;
        if (j0 >= 0 && j0 + 31 < n) {                       // fast contiguous path
            if (j0 + 64 <= i0 + 256) __builtin_prefetch(bp + j0 + 32, 0, 1);
            a.q[0] = *(const uint4*)(ap + j0 + kbA);
            a.q[1] = *(const uint4*)(ap + j0 + kbA + 16);
            b.q[0] = *(const uint4*)(bp + j0 + kbB);
            b.q[1] = *(const uint4*)(bp + j0 + kbB + 16);
        } else {                                            // guarded edge path
#pragma unroll
            for (int e = 0; e < 16; ++e) {
                int ja = j0 + kbA + e + ((e < 8) ? 0 : 8);
                a.s[e] = (ja >= 0 && ja < n) ? ap[ja] : (u16)0;
                int jb = j0 + kbB + e + ((e < 8) ? 0 : 8);
                b.s[e] = (jb >= 0 && jb < n) ? bp[jb] : (u16)0;
            }
        }
        acc = wmma_bf16(a.v, b.v, acc);
    }
    int hb = (lane >= 16) ? 8 : 0;
#pragma unroll
    for (int r = 0; r < 8; ++r)
        Yb[(size_t)(i0 + r + hb) * d + (n0 + (lane & 15))] = f2bf_bits(acc[r]);
}

// ---------------------------------------------------------------------------
// Ysum (+)= LayerNorm(Yo + X)   (torch LN: unbiased std, eps added to std)
// ---------------------------------------------------------------------------
__global__ __launch_bounds__(256)
void resid_ln_acc_kernel(const float* __restrict__ Yo, const float* __restrict__ X,
                         const float* __restrict__ g, const float* __restrict__ be,
                         float* __restrict__ Ysum, int d, int first) {
    __shared__ float r1[256], r2[256];
    int row = blockIdx.x;
    int tid = threadIdx.x;
    const float* yrow = Yo + (size_t)row * d;
    const float* xrow = X + (size_t)row * d;
    float s1 = 0.0f, s2 = 0.0f;
    for (int i = tid; i < d; i += 256) {
        float v = yrow[i] + xrow[i];
        s1 += v; s2 += v * v;
    }
    r1[tid] = s1; r2[tid] = s2;
    __syncthreads();
    for (int off = 128; off > 0; off >>= 1) {
        if (tid < off) { r1[tid] += r1[tid + off]; r2[tid] += r2[tid + off]; }
        __syncthreads();
    }
    float mean = r1[0] / (float)d;
    float var = (r2[0] - (float)d * mean * mean) / (float)(d - 1);
    float inv = 1.0f / (sqrtf(fmaxf(var, 0.0f)) + LN_EPS);
    float* orow = Ysum + (size_t)row * d;
    for (int i = tid; i < d; i += 256) {
        float v = yrow[i] + xrow[i];
        float t = g[i] * (v - mean) * inv + be[i];
        orow[i] = first ? t : (orow[i] + t);
    }
}

// ---------------------------------------------------------------------------
// Head: z = sigmoid( LN(Z3) @ Wd + bd ), one block (128 thr) per row (L3=128)
// ---------------------------------------------------------------------------
__global__ __launch_bounds__(128)
void head_kernel(const float* __restrict__ Z, const float* __restrict__ g,
                 const float* __restrict__ be, const float* __restrict__ Wd,
                 const float* __restrict__ bd, float* __restrict__ out, int L) {
    __shared__ float r1[128], r2[128];
    int row = blockIdx.x;
    int tid = threadIdx.x;
    float v = Z[(size_t)row * L + tid];
    r1[tid] = v; r2[tid] = v * v;
    __syncthreads();
    for (int off = 64; off > 0; off >>= 1) {
        if (tid < off) { r1[tid] += r1[tid + off]; r2[tid] += r2[tid + off]; }
        __syncthreads();
    }
    float mean = r1[0] / (float)L;
    float var = (r2[0] - (float)L * mean * mean) / (float)(L - 1);
    float inv = 1.0f / (sqrtf(fmaxf(var, 0.0f)) + LN_EPS);
    float t = (g[tid] * (v - mean) * inv + be[tid]) * Wd[tid];
    __syncthreads();
    r1[tid] = t;
    __syncthreads();
    for (int off = 64; off > 0; off >>= 1) {
        if (tid < off) r1[tid] += r1[tid + off];
        __syncthreads();
    }
    if (tid == 0) out[row] = 1.0f / (1.0f + expf(-(r1[0] + bd[0])));
}

// ---------------------------------------------------------------------------
extern "C" void kernel_launch(void* const* d_in, const int* in_sizes, int n_in,
                              void* d_out, int out_size, void* d_ws, size_t ws_size,
                              hipStream_t stream) {
    const int n = N_FRAMES, d = D_FEAT;
    const float* x[3] = { (const float*)d_in[0], (const float*)d_in[1], (const float*)d_in[2] };
    const float* Wk = (const float*)d_in[3];
    const float* Wq = (const float*)d_in[4];
    const float* Wv = (const float*)d_in[5];
    const float* Wo = (const float*)d_in[6];
    const float* gy = (const float*)d_in[7];
    const float* by = (const float*)d_in[8];
    const float* Wa = (const float*)d_in[9];
    const float* ba = (const float*)d_in[10];
    const float* Wb = (const float*)d_in[11];
    const float* bb = (const float*)d_in[12];
    const float* Wc = (const float*)d_in[13];
    const float* bc = (const float*)d_in[14];
    const float* gc = (const float*)d_in[15];
    const float* bec = (const float*)d_in[16];
    const float* Wd = (const float*)d_in[17];
    const float* bd = (const float*)d_in[18];

    const size_t ND = (size_t)n * d;          // 4M elems
    char* p = (char*)d_ws;
    auto carve = [&](size_t bytes) { char* r = p; p += (bytes + 255) & ~(size_t)255; return r; };
    u16*   xb   = (u16*)carve(ND * 2);                 // bf16 activations (reused for Ysum)
    u16*   Kb   = (u16*)carve(ND * 2);                 // K bf16 row-major (reused: Z1)
    u16*   Qb   = (u16*)carve(ND * 2);                 // Q bf16 row-major (reused: Z2)
    u16*   Vt   = (u16*)carve(ND * 2);                 // V^T bf16 [d][n]
    u16*   Yb   = (u16*)carve(ND * 2);                 // y bf16 row-major
    float* Yo   = (float*)carve(ND * 4);               // y@Wo fp32 (reused: Z3)
    float* Ysum = (float*)carve(ND * 4);
    u16*   WtB  = (u16*)carve((size_t)n * n * 2);      // w^T bf16 [n][n]
    u16*   WkT  = (u16*)carve((size_t)d * d * 2);
    u16*   WqT  = (u16*)carve((size_t)d * d * 2);
    u16*   WvT  = (u16*)carve((size_t)d * d * 2);
    u16*   WoT  = (u16*)carve((size_t)d * d * 2);
    u16*   WaT  = (u16*)carve((size_t)1024 * 1024 * 2);
    u16*   WbT  = (u16*)carve((size_t)1024 * 512 * 2);
    u16*   WcT  = (u16*)carve((size_t)512 * 128 * 2);
    u16*   Z1b = Kb;       // [n,1024] bf16
    u16*   Z2b = Qb;       // [n,512]  bf16
    float* Z3  = Yo;       // [n,128]  f32

    float* zout = (float*)d_out;              // [n]
    float* attout = zout + n;                 // [n,n] attention of last stream

    (void)hipMemsetAsync(attout, 0, (size_t)n * n * sizeof(float), stream);
    (void)hipMemsetAsync(WtB,    0, (size_t)n * n * sizeof(u16),  stream);

    dim3 blk32(32), blk128(128), blk256(256);
    dim3 gCVT(1024);

    // one-time weight conversions (bf16, transposed to [N][K])
    cvt_bf16_t_kernel<<<gCVT, blk256, 0, stream>>>(Wk, WkT, d, d);
    cvt_bf16_t_kernel<<<gCVT, blk256, 0, stream>>>(Wq, WqT, d, d);
    cvt_bf16_t_kernel<<<gCVT, blk256, 0, stream>>>(Wv, WvT, d, d);
    cvt_bf16_t_kernel<<<gCVT, blk256, 0, stream>>>(Wo, WoT, d, d);
    cvt_bf16_t_kernel<<<gCVT, blk256, 0, stream>>>(Wa, WaT, 1024, 1024);
    cvt_bf16_t_kernel<<<gCVT, blk256, 0, stream>>>(Wb, WbT, 1024, 512);
    cvt_bf16_t_kernel<<<gCVT, blk256, 0, stream>>>(Wc, WcT, 512, 128);

    dim3 gD(d / 64, n / 64);                  // 64x64 GEMM tiles, 4 waves each
    for (int s = 0; s < 3; ++s) {
        cvt_bf16_kernel<<<gCVT, blk256, 0, stream>>>(x[s], xb, ND);
        gemm_bf16_kernel<<<gD, blk128, 0, stream>>>(xb, WkT, nullptr, nullptr, Kb, nullptr, n, d, d, 1.0f,  0);
        gemm_bf16_kernel<<<gD, blk128, 0, stream>>>(xb, WqT, nullptr, nullptr, Qb, nullptr, n, d, d, 0.06f, 0);
        gemm_bf16_kernel<<<gD, blk128, 0, stream>>>(xb, WvT, nullptr, nullptr, nullptr, Vt, n, d, d, 1.0f,  0);

        float* wf = (s == 2) ? attout : nullptr;
        attn_band_kernel<<<dim3(n / 16), blk128, 0, stream>>>(Qb, Kb, wf, WtB, n, d);
        attn_y_kernel<<<dim3(d / 16, n / 16), blk32, 0, stream>>>(WtB, Vt, Yb, n, d);
        gemm_bf16_kernel<<<gD, blk128, 0, stream>>>(Yb, WoT, nullptr, Yo, nullptr, nullptr, n, d, d, 1.0f, 0);
        resid_ln_acc_kernel<<<dim3(n), blk256, 0, stream>>>(Yo, x[s], gy, by, Ysum, d, s == 0 ? 1 : 0);
    }

    // MLP
    cvt_bf16_kernel<<<gCVT, blk256, 0, stream>>>(Ysum, xb, ND);
    gemm_bf16_kernel<<<dim3(1024 / 64, n / 64), blk128, 0, stream>>>(xb,  WaT, ba, nullptr, Z1b, nullptr, n, 1024, 1024, 1.0f, 0);
    gemm_bf16_kernel<<<dim3(512  / 64, n / 64), blk128, 0, stream>>>(Z1b, WbT, bb, nullptr, Z2b, nullptr, n, 512,  1024, 1.0f, 0);
    gemm_bf16_kernel<<<dim3(128  / 64, n / 64), blk128, 0, stream>>>(Z2b, WcT, bc, Z3, nullptr, nullptr,  n, 128,  512,  1.0f, 1);
    head_kernel<<<dim3(n), blk128, 0, stream>>>(Z3, gc, bec, Wd, bd, zout, 128);
}
